// HierarchicalMaskedAttention_3616362463300
// MI455X (gfx1250) — matrix operand order repaired and checked by reading, not compiled, not run on hardware
//
#include <hip/hip_runtime.h>
#include <hip/hip_bf16.h>
#include <math.h>

// ---------------------------------------------------------------------------
// HierarchicalMaskedAttention on gfx1250 (MI455X), bf16 WMMA + fp32 softmax.
//   B=2, S=2048, D=1024, H=16, DH=64, LEVELS=3, FACTOR=2
// Round 3: fixed pointer types for GLOBAL_LOAD_ASYNC_TO_LDS_B128 builtin
// (expects int4 addrspace(1)* / int4 addrspace(3)* per hipcc diagnostic).
// Cooperative async LDS staging for block-shared operands: weight tiles in
// the GEMMs, K/V tiles in attention. Double-buffered, bank-padded.
// ---------------------------------------------------------------------------

#define Bc 2
#define Sc 2048
#define Dc 1024
#define Hc 16
#define DHc 64

typedef __bf16 bf16_t;
typedef __attribute__((ext_vector_type(16))) __bf16 v16bf;
typedef __attribute__((ext_vector_type(8)))  __bf16 v8bf;
typedef __attribute__((ext_vector_type(8)))  float  v8f;
typedef __attribute__((ext_vector_type(4)))  int    v4i;

union V16 { v16bf v; v8bf h[2]; };

#if __has_builtin(__builtin_amdgcn_global_load_async_to_lds_b128)
#define HAVE_ASYNC_LDS 1
#else
#define HAVE_ASYNC_LDS 0
#endif

typedef __attribute__((address_space(1))) v4i* gas_v4i;
typedef __attribute__((address_space(3))) v4i* las_v4i;

// copy 16 bytes global -> LDS (async DMA when available)
__device__ __forceinline__ void copy16_to_lds(void* lds_dst, const void* gsrc) {
#if HAVE_ASYNC_LDS
  __builtin_amdgcn_global_load_async_to_lds_b128((gas_v4i)gsrc, (las_v4i)lds_dst,
                                                 0, 0);
#else
  *(v8bf*)lds_dst = *(const v8bf*)gsrc;
#endif
}

__device__ __forceinline__ void wait_async_lds() {
#if HAVE_ASYNC_LDS
#if __has_builtin(__builtin_amdgcn_s_wait_asynccnt)
  __builtin_amdgcn_s_wait_asynccnt(0);
#else
  asm volatile("s_wait_asynccnt 0x0" ::: "memory");
#endif
#endif
}

__device__ __forceinline__ v8f wmma_bf16f32(v16bf a, v16bf b, v8f c) {
  // D(f32 16x16) = A(bf16 16x32) * B(bf16 32x16) + C
  return __builtin_amdgcn_wmma_f32_16x16x32_bf16(false, a, false, b, (short)0, c,
                                                 false, false);
}

__device__ __forceinline__ v8bf cvt8(const float* p) {
  const float4 x = ((const float4*)p)[0];
  const float4 y = ((const float4*)p)[1];
  v8bf r;
  r[0] = (__bf16)x.x; r[1] = (__bf16)x.y; r[2] = (__bf16)x.z; r[3] = (__bf16)x.w;
  r[4] = (__bf16)y.x; r[5] = (__bf16)y.y; r[6] = (__bf16)y.z; r[7] = (__bf16)y.w;
  return r;
}

// LDS row paddings (elements) for conflict-free 16-lane fragment reads
#define WPAD 40   // 32-elem rows -> 80B stride
#define KPAD 72   // 64-elem rows -> 144B stride

// ---------------------------------------------------------------------------
// 1) Transpose + bf16-convert the four 1024x1024 weights: Wt[w][n][k] = W[k][n]
// ---------------------------------------------------------------------------
__global__ __launch_bounds__(256) void prep_weights_kernel(
    const float* __restrict__ Wq, const float* __restrict__ Wk,
    const float* __restrict__ Wv, const float* __restrict__ Wo,
    bf16_t* __restrict__ Wt) {
  __shared__ float tile[32][33];
  const int wsel = blockIdx.z;
  const float* W = wsel == 0 ? Wq : wsel == 1 ? Wk : wsel == 2 ? Wv : Wo;
  const int k0 = blockIdx.y * 32, n0 = blockIdx.x * 32;
  const int tx = threadIdx.x, ty = threadIdx.y;  // 32 x 8
#pragma unroll
  for (int i = 0; i < 32; i += 8)
    tile[ty + i][tx] = W[(size_t)(k0 + ty + i) * Dc + n0 + tx];
  __syncthreads();
  bf16_t* dst = Wt + ((size_t)wsel * Dc + n0) * Dc + k0;
#pragma unroll
  for (int i = 0; i < 32; i += 8)
    dst[(size_t)(ty + i) * Dc + tx] = (__bf16)tile[tx][ty + i];
}

// ---------------------------------------------------------------------------
// 2) Fused Q/K/V projection GEMM.  M=B*S=4096, N=1024, K=1024.
//    Per wave: 16(M) x 64(N); weight tile staged once per block in LDS.
// ---------------------------------------------------------------------------
__global__ __launch_bounds__(256) void proj_qkv_kernel(
    const float* __restrict__ query, const float* __restrict__ key,
    const float* __restrict__ value, const bf16_t* __restrict__ Wt,
    const float* __restrict__ bq, const float* __restrict__ bk,
    const float* __restrict__ bv, bf16_t* __restrict__ Qh,
    bf16_t* __restrict__ Kh0, bf16_t* __restrict__ Vt0) {
  __shared__ bf16_t LB[2][64][WPAD];   // weight tile: 64 N rows x 32 K
  const int tid = threadIdx.x;
  const int wv = tid >> 5;
  const int lane = tid & 31;
  const int l15 = lane & 15, hi = lane >> 4;
  const int which = blockIdx.z;
  const int m0 = (blockIdx.y * 8 + wv) * 16;
  const int n0 = blockIdx.x * 64;
  const float* X = which == 0 ? query : which == 1 ? key : value;
  const float* bias = which == 0 ? bq : which == 1 ? bk : bv;
  const bf16_t* Wsel = Wt + (size_t)which * Dc * Dc;  // [N][K] bf16
  const float* arow = X + (size_t)(m0 + l15) * Dc + hi * 8;

  // staging map: thread copies one 16B piece; row = tid/4, seg = tid%4
  const int srow = tid >> 2, sseg = (tid & 3) * 8;
  const bf16_t* sbase = Wsel + (size_t)(n0 + srow) * Dc + sseg;

  copy16_to_lds(&LB[0][srow][sseg], sbase);           // k0 = 0
  int cur = 0;
  v8f acc[4] = {};
  for (int k0 = 0; k0 < Dc; k0 += 32) {
    wait_async_lds();
    __syncthreads();
    if (k0 + 32 < Dc)
      copy16_to_lds(&LB[cur ^ 1][srow][sseg], sbase + k0 + 32);

    V16 a;                      // A: 16 rows of X, K = k0..k0+31
    a.h[0] = cvt8(arow + k0);
    a.h[1] = cvt8(arow + k0 + 16);
#pragma unroll
    for (int t = 0; t < 4; t++) {
      V16 bfrag;
      const bf16_t* bp = &LB[cur][t * 16 + l15][hi * 8];
      bfrag.h[0] = *(const v8bf*)bp;
      bfrag.h[1] = *(const v8bf*)(bp + 16);
      acc[t] = wmma_bf16f32(a.v, bfrag.v, acc[t]);
    }
    __syncthreads();
    cur ^= 1;
  }

  const int mb = m0 + hi * 8;
  const int b = mb >> 11;         // / 2048
  const int sbase2 = mb & 2047;
#pragma unroll
  for (int t = 0; t < 4; t++) {
    const int col = n0 + t * 16 + l15;
    const int hh = col >> 6, dh = col & 63;
    const float bb = bias[col];
    if (which == 2) {           // V -> transposed layout, packed 16B store
      v8bf pk;
#pragma unroll
      for (int r = 0; r < 8; r++) pk[r] = (__bf16)(acc[t][r] + bb);
      *(v8bf*)(Vt0 + ((size_t)(b * Hc + hh) * DHc + dh) * Sc + sbase2) = pk;
    } else {
      bf16_t* dst = (which == 0) ? Qh : Kh0;
      const float sc = (which == 0) ? 0.125f : 1.0f;  // 1/sqrt(DH) into Q
#pragma unroll
      for (int r = 0; r < 8; r++)
        dst[((size_t)(b * Hc + hh) * Sc + (sbase2 + r)) * DHc + dh] =
            (__bf16)((acc[t][r] + bb) * sc);
    }
  }
}

// ---------------------------------------------------------------------------
// 3) Mean-pool pairs along sequence for K [bh][S][DH] and Vt [bh*DH][S].
// ---------------------------------------------------------------------------
__global__ __launch_bounds__(256) void downsample_kernel(
    const bf16_t* __restrict__ Kin, bf16_t* __restrict__ Kout,
    const bf16_t* __restrict__ Vin, bf16_t* __restrict__ Vout, int Sin) {
  const int Sout = Sin >> 1;
  const size_t nK = (size_t)Bc * Hc * Sout * DHc;
  const size_t i = (size_t)blockIdx.x * 256 + threadIdx.x;
  if (i < nK) {
    const int d = (int)(i & 63);
    const size_t rest = i >> 6;
    const int s = (int)(rest % Sout);
    const int bh = (int)(rest / Sout);
    const size_t base = ((size_t)bh * Sin + 2 * s) * DHc + d;
    Kout[i] = (__bf16)(0.5f * ((float)Kin[base] + (float)Kin[base + DHc]));
  } else if (i < 2 * nK) {
    const size_t j = i - nK;
    const int s = (int)(j % Sout);
    const size_t row = j / Sout;   // row over bh*DH
    const size_t base = row * Sin + 2 * s;
    Vout[j] = (__bf16)(0.5f * ((float)Vin[base] + (float)Vin[base + 1]));
  }
}

// ---------------------------------------------------------------------------
// 4) Flash attention.  Block = one (b,h) x 8 query tiles; the 8 waves share
//    async-staged K/V chunk tiles in LDS (double buffered).
// ---------------------------------------------------------------------------
__global__ __launch_bounds__(256) void attn_kernel(
    const bf16_t* __restrict__ Qh, const bf16_t* __restrict__ K0,
    const bf16_t* __restrict__ K1, const bf16_t* __restrict__ K2,
    const bf16_t* __restrict__ V0, const bf16_t* __restrict__ V1,
    const bf16_t* __restrict__ V2, const float* __restrict__ w_levels,
    bf16_t* __restrict__ Attn) {
  __shared__ bf16_t LK[2][32][KPAD];   // K chunk: 32 keys x 64 dh
  __shared__ bf16_t LV[2][64][WPAD];   // V chunk: 64 dh x 32 keys
  const int tid = threadIdx.x;
  const int wv = tid >> 5;
  const int lane = tid & 31;
  const int l15 = lane & 15, hi = lane >> 4;
  const int bh = blockIdx.x >> 4;            // 32 (b,h) pairs
  const int qt = (blockIdx.x & 15) * 8 + wv; // 128 query tiles
  const int b = bh >> 4, h = bh & 15;
  const int q0 = qt * 16;
  const float wsum = w_levels[0] + w_levels[1] + w_levels[2];

  // staging maps (one 16B piece per thread per tile)
  const int krow = tid >> 3, kseg = (tid & 7) * 8;   // 32 x 128B
  const int vrow = tid >> 2, vseg = (tid & 3) * 8;   // 64 x 64B

  // Q as B-fragments for S^T = K * Q^T (loaded once; reused for all levels)
  const bf16_t* qbase = Qh + ((size_t)bh * Sc + q0 + l15) * DHc + hi * 8;
  V16 bq0, bq1;
  bq0.h[0] = *(const v8bf*)(qbase); bq0.h[1] = *(const v8bf*)(qbase + 16);   // dh 0..31
  bq1.h[0] = *(const v8bf*)(qbase + 32); bq1.h[1] = *(const v8bf*)(qbase + 48);   // dh 32..63

  const bf16_t* Ks[3] = {K0, K1, K2};
  const bf16_t* Vs[3] = {V0, V1, V2};
  const int Sks[3] = {Sc, Sc / 2, Sc / 4};

  v8f aout[4] = {};                      // weighted sum over levels

  for (int lvl = 0; lvl < 3; lvl++) {
    const int Sk = Sks[lvl];
    const bf16_t* Kl = Ks[lvl] + (size_t)bh * Sk * DHc;   // [Sk][DH]
    const bf16_t* Vl = Vs[lvl] + (size_t)bh * DHc * Sk;   // [DH][Sk]
    float mrun = -INFINITY, lrun = 0.f;
    v8f o[4] = {};

    // prologue stage for chunk j=0
    copy16_to_lds(&LK[0][krow][kseg], Kl + (size_t)krow * DHc + kseg);
    copy16_to_lds(&LV[0][vrow][vseg], Vl + (size_t)vrow * Sk + vseg);
    int cur = 0;

    for (int j = 0; j < Sk; j += 32) {
      wait_async_lds();
      __syncthreads();
      if (j + 32 < Sk) {
        copy16_to_lds(&LK[cur ^ 1][krow][kseg],
                      Kl + (size_t)(j + 32 + krow) * DHc + kseg);
        copy16_to_lds(&LV[cur ^ 1][vrow][vseg],
                      Vl + (size_t)vrow * Sk + (j + 32) + vseg);
      }

      // K A-fragments from LDS (lane owns key row l15 resp. 16+l15)
      const bf16_t* kr0 = &LK[cur][l15][hi * 8];
      const bf16_t* kr1 = &LK[cur][16 + l15][hi * 8];
      V16 ak0a, ak0b, ak1a, ak1b;
      ak0a.h[0] = *(const v8bf*)(kr0);      ak0a.h[1] = *(const v8bf*)(kr0 + 16);
      ak0b.h[0] = *(const v8bf*)(kr0 + 32); ak0b.h[1] = *(const v8bf*)(kr0 + 48);
      ak1a.h[0] = *(const v8bf*)(kr1);      ak1a.h[1] = *(const v8bf*)(kr1 + 16);
      ak1b.h[0] = *(const v8bf*)(kr1 + 32); ak1b.h[1] = *(const v8bf*)(kr1 + 48);

      v8f st0 = {}, st1 = {};               // S^T tiles (keys x queries)
      st0 = wmma_bf16f32(ak0a.v, bq0.v, st0);
      st0 = wmma_bf16f32(ak0b.v, bq1.v, st0);
      st1 = wmma_bf16f32(ak1a.v, bq0.v, st1);
      st1 = wmma_bf16f32(ak1b.v, bq1.v, st1);

      // online softmax (per-lane query = l15; lane pair combined via xor16)
      float cmax = st0[0];
#pragma unroll
      for (int i = 1; i < 8; i++) cmax = fmaxf(cmax, st0[i]);
#pragma unroll
      for (int i = 0; i < 8; i++) cmax = fmaxf(cmax, st1[i]);
      cmax = fmaxf(cmax, __shfl_xor(cmax, 16));
      const float mnew = fmaxf(mrun, cmax);
      const float rescale = __expf(mrun - mnew);   // 0 on first chunk

      V16 ap;                                      // P as A-fragment
      float csum = 0.f;
#pragma unroll
      for (int i = 0; i < 8; i++) {
        const float p0 = __expf(st0[i] - mnew);
        const float p1 = __expf(st1[i] - mnew);
        csum += p0 + p1;
        ap.v[i] = (__bf16)p0;
        ap.v[8 + i] = (__bf16)p1;
      }
      csum += __shfl_xor(csum, 16);
      lrun = lrun * rescale + csum;
      mrun = mnew;

      // rescale O: broadcast per-query factor into the row (VGPR) domain
#pragma unroll
      for (int r = 0; r < 8; r++) {
        const float sr = __shfl(rescale, hi ? (8 + r) : r);
        o[0][r] *= sr; o[1][r] *= sr; o[2][r] *= sr; o[3][r] *= sr;
      }

      // O += P * V  (B-frags from staged transposed V)
#pragma unroll
      for (int t = 0; t < 4; t++) {
        V16 bv;
        const bf16_t* vp = &LV[cur][t * 16 + l15][hi * 8];
        bv.h[0] = *(const v8bf*)vp;
        bv.h[1] = *(const v8bf*)(vp + 16);
        o[t] = wmma_bf16f32(ap.v, bv.v, o[t]);
      }
      __syncthreads();
      cur ^= 1;
    }

    // level epilogue: aout += (w_lvl/wsum) * O / l
    const float wl = w_levels[lvl] / wsum;
    const float inv = wl / lrun;
#pragma unroll
    for (int r = 0; r < 8; r++) {
      const float sr = __shfl(inv, hi ? (8 + r) : r);
      aout[0][r] += sr * o[0][r];
      aout[1][r] += sr * o[1][r];
      aout[2][r] += sr * o[2][r];
      aout[3][r] += sr * o[3][r];
    }
  }

  // store combined heads: Attn[b][s][h*64+dh], bf16
#pragma unroll
  for (int t = 0; t < 4; t++) {
    const int d = h * DHc + t * 16 + l15;
#pragma unroll
    for (int r = 0; r < 8; r++) {
      const int s = q0 + r + hi * 8;
      Attn[((size_t)b * Sc + s) * Dc + d] = (__bf16)aout[t][r];
    }
  }
}

// ---------------------------------------------------------------------------
// 5) Output projection: out = Attn @ Wo + bo  (fp32 out). M=4096, N=K=1024.
// ---------------------------------------------------------------------------
__global__ __launch_bounds__(256) void out_gemm_kernel(
    const bf16_t* __restrict__ Attn, const bf16_t* __restrict__ Wot,
    const float* __restrict__ bo, float* __restrict__ out) {
  __shared__ bf16_t LB[2][64][WPAD];
  const int tid = threadIdx.x;
  const int wv = tid >> 5;
  const int lane = tid & 31;
  const int l15 = lane & 15, hi = lane >> 4;
  const int m0 = (blockIdx.y * 8 + wv) * 16;
  const int n0 = blockIdx.x * 64;
  const bf16_t* arow = Attn + (size_t)(m0 + l15) * Dc + hi * 8;

  const int srow = tid >> 2, sseg = (tid & 3) * 8;
  const bf16_t* sbase = Wot + (size_t)(n0 + srow) * Dc + sseg;

  copy16_to_lds(&LB[0][srow][sseg], sbase);
  int cur = 0;
  v8f acc[4] = {};
  for (int k0 = 0; k0 < Dc; k0 += 32) {
    wait_async_lds();
    __syncthreads();
    if (k0 + 32 < Dc)
      copy16_to_lds(&LB[cur ^ 1][srow][sseg], sbase + k0 + 32);

    V16 a;
    a.h[0] = *(const v8bf*)(arow + k0);
    a.h[1] = *(const v8bf*)(arow + k0 + 16);
#pragma unroll
    for (int t = 0; t < 4; t++) {
      V16 bfrag;
      const bf16_t* bp = &LB[cur][t * 16 + l15][hi * 8];
      bfrag.h[0] = *(const v8bf*)bp;
      bfrag.h[1] = *(const v8bf*)(bp + 16);
      acc[t] = wmma_bf16f32(a.v, bfrag.v, acc[t]);
    }
    __syncthreads();
    cur ^= 1;
  }
#pragma unroll
  for (int t = 0; t < 4; t++) {
    const float bb = bo[n0 + t * 16 + l15];
#pragma unroll
    for (int r = 0; r < 8; r++) {
      const int m = m0 + r + hi * 8;
      out[(size_t)m * Dc + n0 + t * 16 + l15] = acc[t][r] + bb;
    }
  }
}

// ---------------------------------------------------------------------------
// Launch
// ---------------------------------------------------------------------------
extern "C" void kernel_launch(void* const* d_in, const int* in_sizes, int n_in,
                              void* d_out, int out_size, void* d_ws,
                              size_t ws_size, hipStream_t stream) {
  (void)in_sizes; (void)n_in; (void)out_size; (void)ws_size;
  const float* query = (const float*)d_in[0];
  const float* key   = (const float*)d_in[1];
  const float* value = (const float*)d_in[2];
  const float* Wq = (const float*)d_in[3];
  const float* bq = (const float*)d_in[4];
  const float* Wk = (const float*)d_in[5];
  const float* bk = (const float*)d_in[6];
  const float* Wv = (const float*)d_in[7];
  const float* bv = (const float*)d_in[8];
  const float* Wo = (const float*)d_in[9];
  const float* bo = (const float*)d_in[10];
  const float* w_levels = (const float*)d_in[11];
  float* out = (float*)d_out;

  // Workspace carving (~52 MB total)
  char* ws = (char*)d_ws;
  auto take = [&](size_t bytes) {
    char* p = ws;
    ws += (bytes + 255) & ~(size_t)255;
    return p;
  };
  bf16_t* Wt  = (bf16_t*)take((size_t)4 * Dc * Dc * 2);
  bf16_t* Qh  = (bf16_t*)take((size_t)Bc * Hc * Sc * DHc * 2);
  bf16_t* Kh0 = (bf16_t*)take((size_t)Bc * Hc * Sc * DHc * 2);
  bf16_t* Kh1 = (bf16_t*)take((size_t)Bc * Hc * (Sc / 2) * DHc * 2);
  bf16_t* Kh2 = (bf16_t*)take((size_t)Bc * Hc * (Sc / 4) * DHc * 2);
  bf16_t* Vt0 = (bf16_t*)take((size_t)Bc * Hc * Sc * DHc * 2);
  bf16_t* Vt1 = (bf16_t*)take((size_t)Bc * Hc * (Sc / 2) * DHc * 2);
  bf16_t* Vt2 = (bf16_t*)take((size_t)Bc * Hc * (Sc / 4) * DHc * 2);
  bf16_t* Attn = (bf16_t*)take((size_t)Bc * Sc * Dc * 2);

  // 1) weights -> bf16, transposed [w][n][k]
  prep_weights_kernel<<<dim3(Dc / 32, Dc / 32, 4), dim3(32, 8), 0, stream>>>(
      Wq, Wk, Wv, Wo, Wt);

  // 2) Q/K/V projection (z selects matrix)
  proj_qkv_kernel<<<dim3(Dc / 64, (Bc * Sc) / 128, 3), dim3(256), 0, stream>>>(
      query, key, value, Wt, bq, bk, bv, Qh, Kh0, Vt0);

  // 3) downsample level 0 -> 1 -> 2
  {
    size_t n1 = (size_t)2 * Bc * Hc * (Sc / 2) * DHc;
    downsample_kernel<<<dim3((unsigned)((n1 + 255) / 256)), dim3(256), 0,
                        stream>>>(Kh0, Kh1, Vt0, Vt1, Sc);
    size_t n2 = (size_t)2 * Bc * Hc * (Sc / 4) * DHc;
    downsample_kernel<<<dim3((unsigned)((n2 + 255) / 256)), dim3(256), 0,
                        stream>>>(Kh1, Kh2, Vt1, Vt2, Sc / 2);
  }

  // 4) flash attention: block = (b,h) x 8 query tiles, 512 blocks
  attn_kernel<<<dim3(512), dim3(256), 0, stream>>>(
      Qh, Kh0, Kh1, Kh2, Vt0, Vt1, Vt2, w_levels, Attn);

  // 5) output projection (Wo is 4th matrix in Wt)
  out_gemm_kernel<<<dim3(Dc / 64, (Bc * Sc) / 128), dim3(256), 0, stream>>>(
      Attn, Wt + (size_t)3 * Dc * Dc, bo, out);
}
